// DrugGNN_45354854646341
// MI455X (gfx1250) — matrix operand order repaired and checked by reading, not compiled, not run on hardware
//
#include <hip/hip_runtime.h>
#include <hip/hip_bf16.h>

typedef __attribute__((ext_vector_type(16))) __bf16 v16bf;
typedef __attribute__((ext_vector_type(8)))  float  v8f;

#define DD 128

// ---------------- degree / normalization ----------------
__global__ void k_degree(const int* __restrict__ dst, float* __restrict__ deg, int E) {
  int e = blockIdx.x * blockDim.x + threadIdx.x;
  if (e < E) atomicAdd(&deg[dst[e]], 1.0f);
}

__global__ void k_rsqrt(float* __restrict__ dis, int N) {
  int i = blockIdx.x * blockDim.x + threadIdx.x;
  if (i < N) dis[i] = rsqrtf(dis[i] + 1.0f);
}

// ---------------- WMMA GEMM: Y[M x 128] = X[M x 128] @ W[128 x 128] (+bias)(+relu) ----
// bf16x3 split precision (hi*hi + hi*lo + lo*hi, f32 accumulate).
// W is pre-packed into LDS in the exact per-lane B-fragment layout so the hot
// loop uses 16-byte ds_load_b128 accesses; each wave computes TWO 16-row tiles
// so every B fragment feeds 6 WMMAs.
union BU { uint4 u[2]; v16bf v; };

__device__ __forceinline__ void cvt8(const float4 a, const float4 b,
                                     v16bf& hi, v16bf& lo, const int base) {
  float t[8] = {a.x, a.y, a.z, a.w, b.x, b.y, b.z, b.w};
  #pragma unroll
  for (int j = 0; j < 8; ++j) {
    __bf16 h = (__bf16)t[j];
    hi[base + j] = h;
    lo[base + j] = (__bf16)(t[j] - (float)h);
  }
}

__global__ __launch_bounds__(256) void k_gemm(const float* __restrict__ X,
                                              const float* __restrict__ W,
                                              const float* __restrict__ bias,
                                              float* __restrict__ Y,
                                              int M, int relu) {
  // pack[kc][nt][part][lane][8] : part 0,1 = hi (i=0..7, 8..15), part 2,3 = lo
  __shared__ __bf16 pack[4][8][4][32][8];     // 64 KB
  const int tid = threadIdx.x;

  // ---- stage + pre-pack W into per-lane fragment layout ----
  for (int p = tid; p < 64 * 128; p += 256) {
    const int n  = p & 127;
    const int kp = p >> 7;          // 0..63
    const int k  = kp * 2;
    const float w0 = W[(size_t)k * DD + n];
    const float w1 = W[(size_t)(k + 1) * DD + n];
    const __bf16 h0 = (__bf16)w0, h1 = (__bf16)w1;
    const __bf16 g0 = (__bf16)(w0 - (float)h0), g1 = (__bf16)(w1 - (float)h1);
    const int kc = k >> 5, kk = k & 31;
    const int half = (kk >> 3) & 1, i = (kk & 7) + ((kk >> 4) << 3);    // i even
    const int nt = n >> 4, lane = half * 16 + (n & 15);
    const int part = i >> 3, idx = i & 7;
    const unsigned uh = (unsigned)__builtin_bit_cast(unsigned short, h0)
                      | ((unsigned)__builtin_bit_cast(unsigned short, h1) << 16);
    const unsigned ul = (unsigned)__builtin_bit_cast(unsigned short, g0)
                      | ((unsigned)__builtin_bit_cast(unsigned short, g1) << 16);
    *(unsigned*)&pack[kc][nt][part][lane][idx]     = uh;
    *(unsigned*)&pack[kc][nt][part + 2][lane][idx] = ul;
  }
  __syncthreads();

  const int wave = tid >> 5;
  const int lane = tid & 31;
  const int half = lane >> 4;
  const int lo16 = lane & 15;
  const int tile0 = (blockIdx.x * 8 + wave) * 2;
  const int r0 = tile0 * 16;
  if (r0 >= M) return;
  const int r1 = r0 + 16;
  const bool has2 = (r1 < M);

  v8f acc0[8] = {};
  v8f acc1[8] = {};

  const float* xrA = X + (size_t)(r0 + lo16) * DD;
  const float* xrB = has2 ? X + (size_t)(r1 + lo16) * DD : xrA;
  __builtin_prefetch(xrA, 0, 3);   // global_prefetch_b8
  __builtin_prefetch(xrB, 0, 3);

  #pragma unroll
  for (int kc = 0; kc < 4; ++kc) {
    const int k0 = kc * 32;
    // A fragment: elems 0..7 -> k = k0+half*8+i ; elems 8..15 -> k = k0+16+half*8+(i-8)
    const float* pA0 = xrA + k0 + half * 8;
    const float* pA1 = xrA + k0 + 16 + half * 8;
    const float* pB0 = xrB + k0 + half * 8;
    const float* pB1 = xrB + k0 + 16 + half * 8;
    v16bf a0h, a0l, a1h, a1l;
    cvt8(*(const float4*)pA0, *(const float4*)(pA0 + 4), a0h, a0l, 0);
    cvt8(*(const float4*)pA1, *(const float4*)(pA1 + 4), a0h, a0l, 8);
    cvt8(*(const float4*)pB0, *(const float4*)(pB0 + 4), a1h, a1l, 0);
    cvt8(*(const float4*)pB1, *(const float4*)(pB1 + 4), a1h, a1l, 8);

    #pragma unroll
    for (int nt = 0; nt < 8; ++nt) {
      const __bf16* pb = &pack[kc][nt][0][lane][0];
      BU bh, bl;
      bh.u[0] = *(const uint4*)(pb);
      bh.u[1] = *(const uint4*)(pb + 256);   // part stride = 32 lanes * 8 elems
      bl.u[0] = *(const uint4*)(pb + 512);
      bl.u[1] = *(const uint4*)(pb + 768);
      acc0[nt] = __builtin_amdgcn_wmma_f32_16x16x32_bf16(false, a0h, false, bh.v,
                                                         (short)0, acc0[nt], false, false);
      acc0[nt] = __builtin_amdgcn_wmma_f32_16x16x32_bf16(false, a0h, false, bl.v,
                                                         (short)0, acc0[nt], false, false);
      acc0[nt] = __builtin_amdgcn_wmma_f32_16x16x32_bf16(false, a0l, false, bh.v,
                                                         (short)0, acc0[nt], false, false);
      acc1[nt] = __builtin_amdgcn_wmma_f32_16x16x32_bf16(false, a1h, false, bh.v,
                                                         (short)0, acc1[nt], false, false);
      acc1[nt] = __builtin_amdgcn_wmma_f32_16x16x32_bf16(false, a1h, false, bl.v,
                                                         (short)0, acc1[nt], false, false);
      acc1[nt] = __builtin_amdgcn_wmma_f32_16x16x32_bf16(false, a1l, false, bh.v,
                                                         (short)0, acc1[nt], false, false);
    }
  }

  // C/D 16x16 f32 layout: VGPR r, lane L -> M = half*8 + r, N = L%16
  #pragma unroll
  for (int nt = 0; nt < 8; ++nt) {
    const int nn = nt * 16 + lo16;
    const float bv = bias ? bias[nn] : 0.0f;
    #pragma unroll
    for (int r = 0; r < 8; ++r) {
      const int mm = r0 + half * 8 + r;
      float v = acc0[nt][r] + bv;
      if (relu) v = fmaxf(v, 0.0f);
      Y[(size_t)mm * DD + nn] = v;
    }
  }
  if (has2) {
    #pragma unroll
    for (int nt = 0; nt < 8; ++nt) {
      const int nn = nt * 16 + lo16;
      const float bv = bias ? bias[nn] : 0.0f;
      #pragma unroll
      for (int r = 0; r < 8; ++r) {
        const int mm = r1 + half * 8 + r;
        float v = acc1[nt][r] + bv;
        if (relu) v = fmaxf(v, 0.0f);
        Y[(size_t)mm * DD + nn] = v;
      }
    }
  }
}

// ---------------- A = dis^2 * B + b   (self-loop term + bias) ----------------
__global__ void k_selfinit(const float* __restrict__ B, const float* __restrict__ dis,
                           const float* __restrict__ bias, float* __restrict__ A, int N) {
  size_t idx = (size_t)blockIdx.x * blockDim.x + threadIdx.x;
  if (idx >= (size_t)N * DD) return;
  int i = (int)(idx >> 7);
  int c = (int)(idx & 127);
  float d = dis[i];
  A[idx] = d * d * B[idx] + bias[c];
}

// ---------------- edge aggregation: A[dst] += dis[s]*dis[d]*B[src] ----------------
__global__ void k_edge(const int* __restrict__ src, const int* __restrict__ dst,
                       const float* __restrict__ dis, const float* __restrict__ B,
                       float* __restrict__ A, int E) {
  size_t idx = (size_t)blockIdx.x * blockDim.x + threadIdx.x;
  int e = (int)(idx >> 5);
  if (e >= E) return;
  int chunk = (int)(idx & 31);
  int s = src[e], d = dst[e];
  float w = dis[s] * dis[d];
  const float4 v = *reinterpret_cast<const float4*>(B + (size_t)s * DD + chunk * 4);
  float* o = A + (size_t)d * DD + chunk * 4;
  atomicAdd(o + 0, w * v.x);
  atomicAdd(o + 1, w * v.y);
  atomicAdd(o + 2, w * v.z);
  atomicAdd(o + 3, w * v.w);
}

__global__ void k_relu(float* __restrict__ A, size_t n) {
  size_t idx = (size_t)blockIdx.x * blockDim.x + threadIdx.x;
  if (idx < n) A[idx] = fmaxf(A[idx], 0.0f);
}

// ---------------- global mean pool ----------------
__global__ void k_pool(const float* __restrict__ h, const int* __restrict__ batch,
                       float* __restrict__ sums, float* __restrict__ cnts, int N) {
  size_t idx = (size_t)blockIdx.x * blockDim.x + threadIdx.x;
  int i = (int)(idx >> 5);
  if (i >= N) return;
  int chunk = (int)(idx & 31);
  int g = batch[i];
  const float4 v = *reinterpret_cast<const float4*>(h + (size_t)i * DD + chunk * 4);
  float* o = sums + (size_t)g * DD + chunk * 4;
  atomicAdd(o + 0, v.x);
  atomicAdd(o + 1, v.y);
  atomicAdd(o + 2, v.z);
  atomicAdd(o + 3, v.w);
  if (chunk == 0) atomicAdd(&cnts[g], 1.0f);
}

__global__ void k_divide(float* __restrict__ sums, const float* __restrict__ cnts, int total) {
  int idx = blockIdx.x * blockDim.x + threadIdx.x;
  if (idx >= total) return;
  int g = idx >> 7;
  sums[idx] = sums[idx] / fmaxf(cnts[g], 1.0f);
}

static inline int cdiv(long long a, long long b) { return (int)((a + b - 1) / b); }

extern "C" void kernel_launch(void* const* d_in, const int* in_sizes, int n_in,
                              void* d_out, int out_size, void* d_ws, size_t ws_size,
                              hipStream_t stream) {
  (void)n_in; (void)ws_size;
  const float* x  = (const float*)d_in[0];
  const int*   ei = (const int*)d_in[1];
  const int*   batch = (const int*)d_in[2];
  const float* W1 = (const float*)d_in[3]; const float* b1 = (const float*)d_in[4];
  const float* W2 = (const float*)d_in[5]; const float* b2 = (const float*)d_in[6];
  const float* W3 = (const float*)d_in[7]; const float* b3 = (const float*)d_in[8];
  const float* Wp = (const float*)d_in[9]; const float* bp = (const float*)d_in[10];
  float* out = (float*)d_out;

  const int N = in_sizes[0] / DD;
  const int E = in_sizes[1] / 2;
  const int G = out_size / DD;
  const int* src = ei;
  const int* dst = ei + E;

  // workspace layout (floats)
  float* ws   = (float*)d_ws;
  float* bufA = ws;                         // N*128 activations / agg
  float* bufB = bufA + (size_t)N * DD;      // N*128 xw
  float* dis  = bufB + (size_t)N * DD;      // N (deg -> dis in place)
  float* sums = dis + N;                    // G*128
  float* cnts = sums + (size_t)G * DD;      // G

  hipMemsetAsync(dis,  0, (size_t)N * sizeof(float), stream);
  hipMemsetAsync(sums, 0, (size_t)G * DD * sizeof(float), stream);
  hipMemsetAsync(cnts, 0, (size_t)G * sizeof(float), stream);

  k_degree<<<cdiv(E, 256), 256, 0, stream>>>(dst, dis, E);
  k_rsqrt <<<cdiv(N, 256), 256, 0, stream>>>(dis, N);

  const int gemm_blocks = cdiv(N, 256);     // 8 waves * 2 tiles * 16 rows per block
  const size_t ND = (size_t)N * DD;

  const float* Ws[3] = {W1, W2, W3};
  const float* bs[3] = {b1, b2, b3};
  const float* in = x;
  for (int l = 0; l < 3; ++l) {
    k_gemm    <<<gemm_blocks, 256, 0, stream>>>(in, Ws[l], nullptr, bufB, N, 0);
    k_selfinit<<<cdiv((long long)ND, 256), 256, 0, stream>>>(bufB, dis, bs[l], bufA, N);
    k_edge    <<<cdiv((long long)E * 32, 256), 256, 0, stream>>>(src, dst, dis, bufB, bufA, E);
    k_relu    <<<cdiv((long long)ND, 256), 256, 0, stream>>>(bufA, ND);
    in = bufA;
  }

  k_pool  <<<cdiv((long long)N * 32, 256), 256, 0, stream>>>(bufA, batch, sums, cnts, N);
  k_divide<<<cdiv((long long)G * DD, 256), 256, 0, stream>>>(sums, cnts, G * DD);
  // projection head: relu(pooled @ Wp + bp) straight into d_out
  k_gemm  <<<cdiv(G, 256), 256, 0, stream>>>(sums, Wp, bp, out, G, 1);
}
